// MultiHeadSelfAttentionSubLayer_18794776887524
// MI455X (gfx1250) — matrix operand order repaired and checked by reading, not compiled, not run on hardware
//
#include <hip/hip_runtime.h>
#include <hip/hip_bf16.h>

// ---------------------------------------------------------------------------
// MI455X (gfx1250) fused attention + projection.
//   scores = QK^T/32, mask(pad), softmax, O = P V, out = O W + b
// All matmuls on v_wmma_f32_16x16x32_bf16 (wave32). K/V/W converted to bf16
// once (d_ws); O kept in bf16 as the A-matrix of the projection GEMM.
// Workspace layout (needs ~204 MB): Kb | Vb | Wb | Ob  (all bf16)
// ---------------------------------------------------------------------------

typedef __bf16 bf16_t;
typedef __attribute__((ext_vector_type(16))) __bf16        v16bf;
typedef __attribute__((ext_vector_type(4)))  __bf16        v4bf;
typedef __attribute__((ext_vector_type(8)))  float         v8f;
typedef __attribute__((ext_vector_type(4)))  unsigned int  u32x4;

union V16 { v16bf v; u32x4 q[2]; };

static constexpr int   Bc    = 16;
static constexpr int   Sc    = 2048;
static constexpr int   Dc    = 1024;
static constexpr int   Ec    = 1024;
static constexpr float NEGV  = -1.0e13f;    // reference NEG
static constexpr float SCALE = 0.03125f;    // 1/sqrt(1024)

__device__ __forceinline__ v8f wmma_bf16(v16bf a, v16bf b, v8f c) {
  // D = A(16x32 bf16) * B(32x16 bf16) + C(16x16 f32)
  return __builtin_amdgcn_wmma_f32_16x16x32_bf16(false, a, false, b,
                                                 (short)0, c, false, false);
}

// ---------------------------------------------------------------------------
// Kernel 1: f32 -> bf16 bulk convert (float4 in, 4x bf16 out)
// ---------------------------------------------------------------------------
__global__ __launch_bounds__(256)
void attn_cvt_bf16(const float* __restrict__ s, bf16_t* __restrict__ d, int n4) {
  int i  = blockIdx.x * blockDim.x + threadIdx.x;
  int st = gridDim.x * blockDim.x;
  for (; i < n4; i += st) {
    float4 f = ((const float4*)s)[i];
    v4bf o;
    o[0] = (__bf16)f.x; o[1] = (__bf16)f.y;
    o[2] = (__bf16)f.z; o[3] = (__bf16)f.w;
    ((v4bf*)d)[i] = o;
  }
}

// ---------------------------------------------------------------------------
// Kernel 2: flash attention. grid = (S/16, B), block = 256 (8 waves).
// Wave w owns D-slice [w*128, w*128+128). Scores are computed transposed
// (S^T = K_tile * Q^T) so K rows load contiguously as A-fragments and the
// LDS-resident Q^T tile provides reusable B-fragments.
// ---------------------------------------------------------------------------
__global__ __launch_bounds__(256, 2)
void attn_flash(const float* __restrict__ Q,
                const bf16_t* __restrict__ Kb,
                const bf16_t* __restrict__ Vb,
                const long long* __restrict__ pad_idxs,
                bf16_t* __restrict__ Ob) {
  __shared__ __align__(32) __bf16 q_t[Dc * 16];        // Q^T tile: [d][query]
  __shared__ float  sp[8 * 32 * 16];                   // per-wave partial S^T
  __shared__ float  sred[32 * 16];                     // reduced S^T
  __shared__ __align__(16) __bf16 p_lds[16 * 32];      // P tile [query][key]
  __shared__ float  rescale_s[16];
  __shared__ float  invl_s[16];

  const int tid  = threadIdx.x;
  const int lane = tid & 31;
  const int wave = tid >> 5;
  const int wd0  = wave * 128;           // this wave's D-slice base
  const int b    = blockIdx.y;
  const int q0   = blockIdx.x * 16;
  const long long padv = pad_idxs[b];

  // ---- load Q tile -> LDS as Q^T (f32 -> bf16 on the fly) ----
  {
    const float* qb = Q + ((size_t)b * Sc + q0) * Dc;
    const int qi = tid >> 4;             // query 0..15
    const int dl = (tid & 15) * 4;       // d sub-offset
#pragma unroll
    for (int d0 = dl; d0 < Dc; d0 += 64) {
      float4 f = *(const float4*)&qb[(size_t)qi * Dc + d0];
      q_t[(d0 + 0) * 16 + qi] = (__bf16)f.x;
      q_t[(d0 + 1) * 16 + qi] = (__bf16)f.y;
      q_t[(d0 + 2) * 16 + qi] = (__bf16)f.z;
      q_t[(d0 + 3) * 16 + qi] = (__bf16)f.w;
    }
  }
  __syncthreads();

  // ---- preload 4 reusable B-fragments of Q^T (k = d, n = query) ----
  // B layout: lane%16 = N column; 16 bf16 per lane = k rows in table order.
  v16bf bq[4];
#pragma unroll
  for (int t = 0; t < 4; ++t)
#pragma unroll
    for (int s = 0; s < 16; ++s)
      bq[t][s] = q_t[(wd0 + t * 32 + ((lane < 16) ? 0 : 8) + (s < 8 ? s : s + 8)) * 16 + (lane & 15)];

  const bf16_t* Kbase = Kb + (size_t)b * Sc * Dc;
  const bf16_t* Vbase = Vb + (size_t)b * Sc * Dc;

  v8f oacc[8] = {};                      // 16 queries x 128 d (f32)
  float m_run = -3.0e38f, l_run = 0.0f;  // meaningful in tid<16 only
  const int hi8 = (lane >= 16) ? 8 : 0;

  for (int kb = 0; kb < Sc; kb += 32) {
    if (kb + 32 < Sc) {                  // emits global_prefetch_b8
      __builtin_prefetch(&Kbase[(size_t)(kb + 32 + lane) * Dc + wd0], 0, 0);
      __builtin_prefetch(&Vbase[(size_t)(kb + 32 + lane) * Dc + wd0], 0, 0);
    }

    // -- partial S^T over this wave's 128 dims: 2 key sub-tiles x 4 k-steps --
#pragma unroll
    for (int c = 0; c < 2; ++c) {
      v8f sc = {};
      const int key = kb + c * 16 + (lane & 15);
      const bf16_t* kp = Kbase + (size_t)key * Dc;
      const int dl = wd0 + ((lane < 16) ? 0 : 8);   // A 16-bit lane layout
#pragma unroll
      for (int s = 0; s < 4; ++s) {
        V16 a;
        a.q[0] = *(const u32x4*)&kp[dl + s * 32];        // k {0-7}/{8-15}
        a.q[1] = *(const u32x4*)&kp[dl + s * 32 + 16];   // k {16-23}/{24-31}
        sc = wmma_bf16(a.v, bq[s], sc);
      }
#pragma unroll
      for (int r = 0; r < 8; ++r) {      // C layout: lanes<16 M=r, else M=r+8
        const int m = r + hi8;
        sp[wave * 512 + (c * 16 + m) * 16 + (lane & 15)] = sc[r];
      }
    }
    __syncthreads();

    // -- reduce the 8 D-slice partials --
    for (int e = tid; e < 512; e += 256) {
      float acc = sp[e];
#pragma unroll
      for (int w = 1; w < 8; ++w) acc += sp[w * 512 + e];
      sred[e] = acc;
    }
    __syncthreads();

    // -- masked online softmax: lane q of the first 16 threads owns query q --
    if (tid < 16) {
      const int qi = tid;
      const bool qmask = ((long long)(q0 + qi) >= padv);
      float sv[32];
      float mx = m_run;
#pragma unroll
      for (int k = 0; k < 32; ++k) {
        float s = sred[k * 16 + qi] * SCALE;
        if (qmask && ((long long)(kb + k) >= padv)) s = NEGV;
        sv[k] = s;
        mx = fmaxf(mx, s);
      }
      const float rs = __expf(m_run - mx);
      float ls = l_run * rs;
#pragma unroll
      for (int k = 0; k < 32; ++k) {
        const float p = __expf(sv[k] - mx);
        ls += p;
        p_lds[qi * 32 + k] = (__bf16)p;
      }
      m_run = mx; l_run = ls;
      rescale_s[qi] = rs;
      invl_s[qi]    = 1.0f / ls;         // final iteration's value is used
    }
    __syncthreads();

    // -- rescale O and accumulate P * V --
    float rsv[8];
#pragma unroll
    for (int r = 0; r < 8; ++r) rsv[r] = rescale_s[r + hi8];

    // A fragment of P (M = query, k = 32 keys), broadcast from LDS
    V16 ap;
    {
      const int qrow = lane & 15;
      const int koff = (lane < 16) ? 0 : 8;
      ap.q[0] = *(const u32x4*)&p_lds[qrow * 32 + koff];
      ap.q[1] = *(const u32x4*)&p_lds[qrow * 32 + koff + 16];
    }
    const bf16_t* vp = Vbase + (size_t)kb * Dc + wd0 + (lane & 15);  // lane%16 = d column
#pragma unroll
    for (int t = 0; t < 8; ++t) {
#pragma unroll
      for (int r = 0; r < 8; ++r) oacc[t][r] *= rsv[r];
      v16bf bv;  // B: lane%16 = d column, 16 keys in table order
#pragma unroll
      for (int s = 0; s < 16; ++s)
        bv[s] = vp[(size_t)(hi8 + (s < 8 ? s : s + 8)) * Dc + t * 16];
      oacc[t] = wmma_bf16(ap.v, bv, oacc[t]);
    }
  }

  // ---- normalize and store O slice as bf16 ----
  float ilv[8];
#pragma unroll
  for (int r = 0; r < 8; ++r) ilv[r] = invl_s[r + hi8];
  bf16_t* ob = Ob + ((size_t)b * Sc + q0) * Dc + wd0;
#pragma unroll
  for (int t = 0; t < 8; ++t)
#pragma unroll
    for (int r = 0; r < 8; ++r) {
      const int qrow = r + hi8;
      ob[(size_t)qrow * Dc + t * 16 + (lane & 15)] = (__bf16)(oacc[t][r] * ilv[r]);
    }
}

// ---------------------------------------------------------------------------
// Kernel 3: out = Ob(32768 x 1024) * Wb(1024 x 1024) + bias.
// Block tile 128x128, 8 waves each 32x64, K-step 32, LDS single-buffered tiles.
// ---------------------------------------------------------------------------
__global__ __launch_bounds__(256, 2)
void attn_proj_gemm(const bf16_t* __restrict__ A,
                    const bf16_t* __restrict__ Wb,
                    const float* __restrict__ bias,
                    float* __restrict__ out) {
  __shared__ __align__(32) __bf16 a_lds[128 * 32];
  __shared__ __align__(32) __bf16 b_lds[32 * 128];

  const int tid  = threadIdx.x;
  const int lane = tid & 31;
  const int wave = tid >> 5;
  const int m0   = blockIdx.y * 128;
  const int n0   = blockIdx.x * 128;
  const int wm   = wave & 3;             // 4 M-slices of 32
  const int wn   = wave >> 2;            // 2 N-slices of 64
  const int hi8  = (lane >= 16) ? 8 : 0;

  v8f acc[2][4] = {};

  for (int kk = 0; kk < Dc; kk += 32) {
    {
      const int ar = tid >> 1, ah = tid & 1;
      *(v16bf*)&a_lds[ar * 32 + ah * 16] =
          *(const v16bf*)&A[(size_t)(m0 + ar) * Dc + kk + ah * 16];
      const int br = tid >> 3, bs = tid & 7;
      *(v16bf*)&b_lds[br * 128 + bs * 16] =
          *(const v16bf*)&Wb[(size_t)(kk + br) * Ec + n0 + bs * 16];
    }
    __syncthreads();

    V16 am[2];
    {
      const int koff = (lane < 16) ? 0 : 8;
      const int qrow = lane & 15;
#pragma unroll
      for (int mi = 0; mi < 2; ++mi) {
        const int ar = wm * 32 + mi * 16 + qrow;
        am[mi].q[0] = *(const u32x4*)&a_lds[ar * 32 + koff];
        am[mi].q[1] = *(const u32x4*)&a_lds[ar * 32 + koff + 16];
      }
    }
#pragma unroll
    for (int ni = 0; ni < 4; ++ni) {
      v16bf bt;
#pragma unroll
      for (int s = 0; s < 16; ++s)
        bt[s] = b_lds[(hi8 + (s < 8 ? s : s + 8)) * 128 + wn * 64 + ni * 16 + (lane & 15)];
#pragma unroll
      for (int mi = 0; mi < 2; ++mi)
        acc[mi][ni] = wmma_bf16(am[mi].v, bt, acc[mi][ni]);
    }
    __syncthreads();
  }

#pragma unroll
  for (int ni = 0; ni < 4; ++ni) {
    const int col = n0 + wn * 64 + ni * 16 + (lane & 15);
    const float bv = bias[col];
#pragma unroll
    for (int mi = 0; mi < 2; ++mi)
#pragma unroll
      for (int r = 0; r < 8; ++r) {
        const int row = m0 + wm * 32 + mi * 16 + r + hi8;
        out[(size_t)row * Ec + col] = acc[mi][ni][r] + bv;
      }
  }
}

// ---------------------------------------------------------------------------
extern "C" void kernel_launch(void* const* d_in, const int* in_sizes, int n_in,
                              void* d_out, int out_size, void* d_ws, size_t ws_size,
                              hipStream_t stream) {
  const float*      Q   = (const float*)d_in[0];
  const float*      K   = (const float*)d_in[1];
  const float*      V   = (const float*)d_in[2];
  const long long*  pad = (const long long*)d_in[3];   // jnp.int64
  const float*      W   = (const float*)d_in[4];
  const float*      b   = (const float*)d_in[5];
  float*            out = (float*)d_out;

  // workspace: Kb | Vb | Wb | Ob  (bf16) -> ~204 MB
  bf16_t* Kb = (bf16_t*)d_ws;
  bf16_t* Vb = Kb + (size_t)Bc * Sc * Dc;
  bf16_t* Wb = Vb + (size_t)Bc * Sc * Dc;
  bf16_t* Ob = Wb + (size_t)Dc * Ec;

  const int n4_kv = (Bc * Sc * Dc) / 4;   // 8,388,608
  const int n4_w  = (Dc * Ec) / 4;        // 262,144

  attn_cvt_bf16<<<dim3(4096), 256, 0, stream>>>(K, Kb, n4_kv);
  attn_cvt_bf16<<<dim3(4096), 256, 0, stream>>>(V, Vb, n4_kv);
  attn_cvt_bf16<<<dim3(1024), 256, 0, stream>>>(W, Wb, n4_w);

  attn_flash<<<dim3(Sc / 16, Bc), 256, 0, stream>>>(Q, Kb, Vb, pad, Ob);

  attn_proj_gemm<<<dim3(Ec / 128, (Bc * Sc) / 128), 256, 0, stream>>>(Ob, Wb, b, out);
}
